// InvariantPointAttention_13168369729874
// MI455X (gfx1250) — matrix operand order repaired and checked by reading, not compiled, not run on hardware
//
#include <hip/hip_runtime.h>
#include <hip/hip_bf16.h>
#include <math.h>

// ---------------------------------------------------------------------------
// Invariant Point Attention for MI455X (gfx1250), wave32, bf16 WMMA.
// K/KA/VC/Q tensors are stored in WMMA-fragment order so LDS staging is a
// flat copy (async-to-LDS when available) and fragment reads are b128 loads.
// ---------------------------------------------------------------------------

typedef __attribute__((ext_vector_type(16))) __bf16 v16bf;
typedef __attribute__((ext_vector_type(8)))  float  v8f;
typedef __attribute__((ext_vector_type(4)))  int    v4i;

#define DEV __device__ __forceinline__

#if defined(__AMDGCN__) && \
    __has_builtin(__builtin_amdgcn_global_load_async_to_lds_b128) && \
    __has_builtin(__builtin_amdgcn_s_wait_asynccnt)
#define USE_ASYNC_LDS 1
typedef __attribute__((address_space(1))) v4i* gptr_b128;
typedef __attribute__((address_space(3))) v4i* lptr_b128;
#else
#define USE_ASYNC_LDS 0
#endif

constexpr int N_   = 2048;
constexpr int H_   = 8;
constexpr int C_   = 128;
constexpr int PQ_  = 8;
constexpr int PV_  = 12;
constexpr int R_   = 2;
constexpr int CS_  = 256;
constexpr int CZ_  = 128;
constexpr int CZ4_ = 32;
constexpr int HC_  = H_ * C_;              // 1024
constexpr int VCW  = 240;                  // v(128) | vp(48, 36 used) | z2d(64)
constexpr int NT   = VCW / 16;             // 15 output col tiles
constexpr int FEATW = HC_ + 4 * H_ * PV_ + H_ * CZ4_;  // 1664

// ---- workspace layout (bytes) ----
constexpr size_t OFF_QF   = 0;
constexpr size_t OFF_KVF  = OFF_QF   + (size_t)N_ * HC_ * 4;
constexpr size_t OFF_QPF  = OFF_KVF  + (size_t)N_ * 2 * HC_ * 4;
constexpr size_t OFF_KVPF = OFF_QPF  + (size_t)N_ * H_ * PQ_ * 3 * 4;
constexpr size_t OFF_B1F  = OFF_KVPF + (size_t)N_ * H_ * (PQ_ + PV_) * 3 * 4;
constexpr size_t OFF_B2F  = OFF_B1F  + (size_t)N_ * R_ * H_ * 4;
constexpr size_t OFF_Z1DF = OFF_B2F  + (size_t)N_ * R_ * H_ * 4;
constexpr size_t OFF_Z2DF = OFF_Z1DF + (size_t)N_ * R_ * CZ4_ * 4;
constexpr size_t OFF_QBF  = OFF_Z2DF + (size_t)N_ * R_ * CZ4_ * 4;
constexpr size_t OFF_KBF  = OFF_QBF  + (size_t)H_ * N_ * C_ * 2;
constexpr size_t OFF_QABF = OFF_KBF  + (size_t)H_ * N_ * C_ * 2;
constexpr size_t OFF_KABF = OFF_QABF + (size_t)H_ * N_ * 32 * 2;
constexpr size_t OFF_VCBF = OFF_KABF + (size_t)H_ * N_ * 32 * 2;
constexpr size_t OFF_OACC = OFF_VCBF + (size_t)H_ * N_ * VCW * 2;
constexpr size_t OFF_FEAT = OFF_OACC + (size_t)H_ * N_ * VCW * 4;

DEV v8f wmma_bf16(v16bf a, v16bf b, v8f c) {
    return __builtin_amdgcn_wmma_f32_16x16x32_bf16(
        false, a, false, b, (short)0, c, false, false);
}

// ---------------------------------------------------------------------------
// Fragment-order indexing (ISA 7.12.2 VGPR layouts, wave32).
// A fragment (16 rows i x 32 cols k): lane = ((k&15)>>3)*16 + i,
//   slot = (k&7) | ((k&16)>>1). 16 contiguous bf16 per lane.
// B fragment (32 rows k x 16 cols n): lane = ((k&15)>>3)*16 + n, slot = (k&7) | ((k&16)>>1).
// ---------------------------------------------------------------------------
DEV int AOFF(int i, int k)  { return ((((k & 15) >> 3) << 4) + i) * 16 + ((k & 7) | ((k & 16) >> 1)); }
DEV int BOFF(int k, int n)  { return ((((k & 15) >> 3) << 4) + n) * 16 + ((k & 7) | ((k & 16) >> 1)); }

// qbf : [H][N/16][4 kwin][512]   (A-order, rows = i)
DEV size_t QIDX (int h, int i, int c) {
    return ((((size_t)h * (N_/16) + (i >> 4)) * 4 + (c >> 5)) << 9) + AOFF(i & 15, c & 31);
}
// qabf: [H][N/16][512]
DEV size_t QAIDX(int h, int i, int c) {
    return (((size_t)h * (N_/16) + (i >> 4)) << 9) + AOFF(i & 15, c);
}
// kbf : [H][N/16][4 kwin][512]   (B-order, cols = j, rows = channel)
DEV size_t KIDX (int h, int j, int c) {
    return ((((size_t)h * (N_/16) + (j >> 4)) * 4 + (c >> 5)) << 9) + BOFF(c & 31, j & 15);
}
// kabf: [H][N/16][512]
DEV size_t KAIDX(int h, int j, int c) {
    return (((size_t)h * (N_/16) + (j >> 4)) << 9) + BOFF(c, j & 15);
}
// vcbf: [H][N/32][15 coltile][512]  (B-order, rows = j(32), cols = feature)
DEV size_t VCIDX(int h, int j, int c) {
    return ((((size_t)h * (N_/32) + (j >> 5)) * 15 + (c >> 4)) << 9) + BOFF(j & 31, c & 15);
}

// ---------------------------------------------------------------------------
// Generic GEMM:  Y[nrows x M] = X[nrows x K] @ W[M x K]^T + bias  (f32 io)
// ---------------------------------------------------------------------------
__global__ void gemm_xwT_bf16(const float* __restrict__ X,
                              const float* __restrict__ W,
                              const float* __restrict__ bias,
                              float* __restrict__ Y,
                              int nrows, int K, int M)
{
    const int lane = threadIdx.x & 31;
    const int wave = threadIdx.x >> 5;
    const int ln   = lane & 15;
    const int kg8  = (lane >> 4) << 3;
    const int kg16 = (lane >> 4) << 4;
    const int i0   = blockIdx.y * 64 + wave * 16;
    const int n0   = blockIdx.x * 64;

    v8f acc[4];
    #pragma unroll
    for (int t = 0; t < 4; ++t)
        #pragma unroll
        for (int r = 0; r < 8; ++r) acc[t][r] = 0.0f;

    const float* xrow = X + (size_t)(i0 + ln) * K;

    for (int kk = 0; kk < K; kk += 32) {
        v16bf a;
        #pragma unroll
        for (int v = 0; v < 8; ++v) {
            int k = kk + kg16 + 2 * v;
            a[2 * v]     = (__bf16)xrow[k];
            a[2 * v + 1] = (__bf16)xrow[k + 1];
        }
        #pragma unroll
        for (int t = 0; t < 4; ++t) {
            int n = n0 + t * 16 + ln;
            if (n >= M) n = M - 1;
            const float* wrow = W + (size_t)n * K + kk + kg16;
            v16bf b;
            #pragma unroll
            for (int v = 0; v < 8; ++v) {
                b[2 * v]     = (__bf16)wrow[2 * v];
                b[2 * v + 1] = (__bf16)wrow[2 * v + 1];
            }
            acc[t] = wmma_bf16(a, b, acc[t]);
        }
    }

    #pragma unroll
    for (int r = 0; r < 8; ++r) {
        int row = i0 + r + ((lane >> 4) << 3);
        #pragma unroll
        for (int t = 0; t < 4; ++t) {
            int col = n0 + t * 16 + ln;
            if (col < M)
                Y[(size_t)row * M + col] = acc[t][r] + bias[col];
        }
    }
}

// ---------------------------------------------------------------------------
// Pack kernel: rotations + augmented features, scatter into fragment order.
// ---------------------------------------------------------------------------
__global__ void pack_kernel(const float* __restrict__ qf,
                            const float* __restrict__ kvf,
                            const float* __restrict__ qpf,
                            const float* __restrict__ kvpf,
                            const float* __restrict__ b1f,
                            const float* __restrict__ b2f,
                            const float* __restrict__ z2df,
                            const float* __restrict__ rotp,
                            const float* __restrict__ transp,
                            const unsigned char* __restrict__ maskp,
                            const float* __restrict__ hwp,
                            __bf16* __restrict__ qbf,
                            __bf16* __restrict__ kbf,
                            __bf16* __restrict__ qabf,
                            __bf16* __restrict__ kabf,
                            __bf16* __restrict__ vcbf)
{
    const int i   = blockIdx.x;
    const int tid = threadIdx.x;            // 128 threads

    __shared__ float qsq[H_ * PQ_];
    __shared__ float ksq[H_ * PQ_];

    const float* Rm = rotp + (size_t)i * 9;
    const float r00 = Rm[0], r01 = Rm[1], r02 = Rm[2];
    const float r10 = Rm[3], r11 = Rm[4], r12 = Rm[5];
    const float r20 = Rm[6], r21 = Rm[7], r22 = Rm[8];
    const float tx = transp[i * 3 + 0], ty = transp[i * 3 + 1], tz = transp[i * 3 + 2];
    const float hwscale = rsqrtf(108.0f);
    const float qscale  = rsqrtf(3.0f * (float)C_);

    for (int idx = tid; idx < H_ * PQ_; idx += 128) {
        int h = idx >> 3, p = idx & 7;
        const float* pt = qpf + (size_t)i * (H_ * PQ_ * 3) + idx * 3;
        float x = pt[0], y = pt[1], z = pt[2];
        float gx = r00 * x + r01 * y + r02 * z + tx;
        float gy = r10 * x + r11 * y + r12 * z + ty;
        float gz = r20 * x + r21 * y + r22 * z + tz;
        qsq[idx] = gx * gx + gy * gy + gz * gz;
        float hw = log1pf(__expf(hwp[h])) * hwscale;
        qabf[QAIDX(h, i, p * 3 + 0)] = (__bf16)(hw * gx);
        qabf[QAIDX(h, i, p * 3 + 1)] = (__bf16)(hw * gy);
        qabf[QAIDX(h, i, p * 3 + 2)] = (__bf16)(hw * gz);
    }
    for (int idx = tid; idx < H_ * (PQ_ + PV_); idx += 128) {
        int h = idx / (PQ_ + PV_), p = idx % (PQ_ + PV_);
        const float* pt = kvpf + (size_t)i * (H_ * (PQ_ + PV_) * 3) + idx * 3;
        float x = pt[0], y = pt[1], z = pt[2];
        float gx = r00 * x + r01 * y + r02 * z + tx;
        float gy = r10 * x + r11 * y + r12 * z + ty;
        float gz = r20 * x + r21 * y + r22 * z + tz;
        if (p < PQ_) {
            ksq[h * PQ_ + p] = gx * gx + gy * gy + gz * gz;
            kabf[KAIDX(h, i, p * 3 + 0)] = (__bf16)gx;
            kabf[KAIDX(h, i, p * 3 + 1)] = (__bf16)gy;
            kabf[KAIDX(h, i, p * 3 + 2)] = (__bf16)gz;
        } else {
            int c = 128 + (p - PQ_) * 3;
            vcbf[VCIDX(h, i, c + 0)] = (__bf16)gx;
            vcbf[VCIDX(h, i, c + 1)] = (__bf16)gy;
            vcbf[VCIDX(h, i, c + 2)] = (__bf16)gz;
        }
    }
    __syncthreads();

    for (int h = tid; h < H_; h += 128) {
        float qn = 0.f, kn = 0.f;
        #pragma unroll
        for (int p = 0; p < PQ_; ++p) { qn += qsq[h * PQ_ + p]; kn += ksq[h * PQ_ + p]; }
        float hw = log1pf(__expf(hwp[h])) * hwscale;
        qabf[QAIDX(h, i, 24)] = (__bf16)(-0.5f * hw * qn);
        qabf[QAIDX(h, i, 25)] = (__bf16)1.0f;
        qabf[QAIDX(h, i, 26)] = (__bf16)b1f[(size_t)(i * 2 + 0) * H_ + h];
        qabf[QAIDX(h, i, 27)] = (__bf16)b1f[(size_t)(i * 2 + 1) * H_ + h];
        qabf[QAIDX(h, i, 28)] = (__bf16)1.0f;
        qabf[QAIDX(h, i, 29)] = (__bf16)0.f;
        qabf[QAIDX(h, i, 30)] = (__bf16)0.f;
        qabf[QAIDX(h, i, 31)] = (__bf16)0.f;
        kabf[KAIDX(h, i, 24)] = (__bf16)1.0f;
        kabf[KAIDX(h, i, 25)] = (__bf16)(-0.5f * hw * kn);
        kabf[KAIDX(h, i, 26)] = (__bf16)b2f[(size_t)(i * 2 + 0) * H_ + h];
        kabf[KAIDX(h, i, 27)] = (__bf16)b2f[(size_t)(i * 2 + 1) * H_ + h];
        kabf[KAIDX(h, i, 28)] = (__bf16)(maskp[i] ? 0.0f : -100000.0f);
        kabf[KAIDX(h, i, 29)] = (__bf16)0.f;
        kabf[KAIDX(h, i, 30)] = (__bf16)0.f;
        kabf[KAIDX(h, i, 31)] = (__bf16)0.f;
    }

    for (int idx = tid; idx < HC_; idx += 128) {
        int h = idx >> 7, c = idx & 127;
        qbf[QIDX(h, i, c)]  = (__bf16)(qf[(size_t)i * HC_ + idx] * qscale);
        kbf[KIDX(h, i, c)]  = (__bf16)kvf[(size_t)i * 2 * HC_ + h * 2 * C_ + c];
        vcbf[VCIDX(h, i, c)] = (__bf16)kvf[(size_t)i * 2 * HC_ + h * 2 * C_ + C_ + c];
    }
    for (int idx = tid; idx < H_ * 12; idx += 128) {
        int h = idx / 12, j = idx % 12;
        vcbf[VCIDX(h, i, 164 + j)] = (__bf16)0.0f;
    }
    for (int idx = tid; idx < H_ * R_ * CZ4_; idx += 128) {
        int h = idx >> 6, rc = idx & 63;
        vcbf[VCIDX(h, i, 176 + rc)] =
            (__bf16)z2df[((size_t)i * R_ + (rc >> 5)) * CZ4_ + (rc & 31)];
    }
}

// ---------------------------------------------------------------------------
// Fused flash attention. Block = 4 waves = 64 query rows, per head.
// Tiles arrive in fragment order: staging is a flat (async) copy and every
// fragment read is one 32B LDS access.
// ---------------------------------------------------------------------------
__global__ void __launch_bounds__(128)
flash_ipa_kernel(const __bf16* __restrict__ qbf,
                 const __bf16* __restrict__ kbf,
                 const __bf16* __restrict__ qabf,
                 const __bf16* __restrict__ kabf,
                 const __bf16* __restrict__ vcbf,
                 float* __restrict__ oacc)
{
    const int lane = threadIdx.x & 31;
    const int wave = threadIdx.x >> 5;
    const int tid  = threadIdx.x;
    const int ln   = lane & 15;
    const int kg8  = (lane >> 4) << 3;
    const int h    = blockIdx.y;
    const int i0   = blockIdx.x * 64 + wave * 16;

    __shared__ __align__(32) __bf16 KtL [2 * 4 * 512];  // [jn][kwin][frag512]
    __shared__ __align__(32) __bf16 KAtL[2 * 512];      // [jn][frag512]
    __shared__ __align__(32) __bf16 VCtL[NT * 512];     // [coltile][frag512]
    __shared__ __align__(32) __bf16 PfL [4 * 512];      // per-wave P fragment

    const __bf16* ksrc0  = kbf  + ((size_t)h * (N_/16)) * 4 * 512;
    const __bf16* kasrc0 = kabf + ((size_t)h * (N_/16)) * 512;
    const __bf16* vsrc0  = vcbf + ((size_t)h * (N_/32)) * NT * 512;

    // ---- preload Q fragments (A order -> contiguous per lane) ----
    const __bf16* qsrc = qbf + (((size_t)h * (N_/16) + (i0 >> 4)) * 4) * 512;
    v16bf qfrag[4];
    #pragma unroll
    for (int kk = 0; kk < 4; ++kk)
        qfrag[kk] = *reinterpret_cast<const v16bf*>(qsrc + kk * 512 + lane * 16);
    v16bf qafrag = *reinterpret_cast<const v16bf*>(
        qabf + (((size_t)h * (N_/16) + (i0 >> 4)) << 9) + lane * 16);

    v8f accv[NT];
    #pragma unroll
    for (int t = 0; t < NT; ++t)
        #pragma unroll
        for (int r = 0; r < 8; ++r) accv[t][r] = 0.0f;
    float mrow[8], lrow[8];
    #pragma unroll
    for (int r = 0; r < 8; ++r) { mrow[r] = -3.0e38f; lrow[r] = 0.0f; }

    for (int jt = 0; jt < N_ / 32; ++jt) {
        const __bf16* ksrc  = ksrc0  + (size_t)(jt * 2) * 4 * 512;  // 4096 bf16
        const __bf16* kasrc = kasrc0 + (size_t)(jt * 2) * 512;      // 1024 bf16
        const __bf16* vsrc  = vsrc0  + (size_t)jt * NT * 512;       // 7680 bf16

        __syncthreads();   // previous iteration's tile reads complete

#if USE_ASYNC_LDS
        {   // CDNA5 async global->LDS staging (ASYNCcnt)
            const char* gk = (const char*)ksrc;
            const char* ga = (const char*)kasrc;
            const char* gv = (const char*)vsrc;
            char* lk = (char*)KtL;
            char* la = (char*)KAtL;
            char* lv = (char*)VCtL;
            for (int idx = tid; idx < 512; idx += 128)
                __builtin_amdgcn_global_load_async_to_lds_b128(
                    (gptr_b128)(gk + idx * 16),
                    (lptr_b128)(lk + idx * 16), 0, 0);
            for (int idx = tid; idx < 128; idx += 128)
                __builtin_amdgcn_global_load_async_to_lds_b128(
                    (gptr_b128)(ga + idx * 16),
                    (lptr_b128)(la + idx * 16), 0, 0);
            for (int idx = tid; idx < 960; idx += 128)
                __builtin_amdgcn_global_load_async_to_lds_b128(
                    (gptr_b128)(gv + idx * 16),
                    (lptr_b128)(lv + idx * 16), 0, 0);
        }
        __builtin_amdgcn_s_wait_asynccnt(0);
#else
        {
            const uint32_t* s0 = (const uint32_t*)ksrc;
            uint32_t* d0 = (uint32_t*)KtL;
            for (int idx = tid; idx < 2048; idx += 128) d0[idx] = s0[idx];
            const uint32_t* s1 = (const uint32_t*)kasrc;
            uint32_t* d1 = (uint32_t*)KAtL;
            for (int idx = tid; idx < 512; idx += 128) d1[idx] = s1[idx];
            const uint32_t* s2 = (const uint32_t*)vsrc;
            uint32_t* d2 = (uint32_t*)VCtL;
            for (int idx = tid; idx < 3840; idx += 128) d2[idx] = s2[idx];
        }
#endif
        if (jt + 1 < N_ / 32) {   // prefetch next tiles into L2
            __builtin_prefetch(ksrc + 4096, 0, 0);
            __builtin_prefetch(vsrc + NT * 512, 0, 0);
        }
        __syncthreads();

        // ---- logits: keys [j0, j0+16) and [j0+16, j0+32) ----
        v8f s0, s1;
        #pragma unroll
        for (int r = 0; r < 8; ++r) { s0[r] = 0.0f; s1[r] = 0.0f; }
        #pragma unroll
        for (int kk = 0; kk < 4; ++kk) {
            v16bf b0 = *reinterpret_cast<const v16bf*>(&KtL[(0 * 4 + kk) * 512 + lane * 16]);
            v16bf b1 = *reinterpret_cast<const v16bf*>(&KtL[(1 * 4 + kk) * 512 + lane * 16]);
            s0 = wmma_bf16(qfrag[kk], b0, s0);
            s1 = wmma_bf16(qfrag[kk], b1, s1);
        }
        {
            v16bf b0 = *reinterpret_cast<const v16bf*>(&KAtL[lane * 16]);
            v16bf b1 = *reinterpret_cast<const v16bf*>(&KAtL[512 + lane * 16]);
            s0 = wmma_bf16(qafrag, b0, s0);
            s1 = wmma_bf16(qafrag, b1, s1);
        }

        // ---- online softmax; scatter P directly into A-fragment slots ----
        #pragma unroll
        for (int r = 0; r < 8; ++r) {
            float a0 = s0[r], a1 = s1[r];
            float mx = fmaxf(a0, a1);
            mx = fmaxf(mx, __shfl_xor(mx, 1));
            mx = fmaxf(mx, __shfl_xor(mx, 2));
            mx = fmaxf(mx, __shfl_xor(mx, 4));
            mx = fmaxf(mx, __shfl_xor(mx, 8));
            float mnew = fmaxf(mrow[r], mx);
            float sc   = __expf(mrow[r] - mnew);
            float p0   = __expf(a0 - mnew);
            float p1   = __expf(a1 - mnew);
            float sm = p0 + p1;
            sm += __shfl_xor(sm, 1);
            sm += __shfl_xor(sm, 2);
            sm += __shfl_xor(sm, 4);
            sm += __shfl_xor(sm, 8);
            lrow[r] = lrow[r] * sc + sm;
            mrow[r] = mnew;
            #pragma unroll
            for (int t = 0; t < NT; ++t) accv[t][r] *= sc;
            int prow = r + kg8;                       // this lane's D-layout row
            int dl   = ((ln >> 3) << 4) + prow;       // A-frag dest lane
            PfL[wave * 512 + dl * 16 + (ln & 7)]     = (__bf16)p0;   // j = ln
            PfL[wave * 512 + dl * 16 + (ln & 7) + 8] = (__bf16)p1;   // j = 16+ln
        }
        asm volatile("" ::: "memory");   // order DS store -> DS load (same wave)

        v16bf pf = *reinterpret_cast<const v16bf*>(&PfL[wave * 512 + lane * 16]);

        // ---- accumulate P @ [v | vp | z2d] ----
        #pragma unroll
        for (int t = 0; t < NT; ++t) {
            v16bf b = *reinterpret_cast<const v16bf*>(&VCtL[t * 512 + lane * 16]);
            accv[t] = wmma_bf16(pf, b, accv[t]);
        }
    }

    // ---- epilogue: normalize and store [H][N][240] ----
    #pragma unroll
    for (int r = 0; r < 8; ++r) {
        float inv = 1.0f / fmaxf(lrow[r], 1e-20f);
        int row = i0 + r + kg8;
        #pragma unroll
        for (int t = 0; t < NT; ++t)
            oacc[((size_t)h * N_ + row) * VCW + t * 16 + ln] = accv[t][r] * inv;
    }
}

// ---------------------------------------------------------------------------
// Finalize: inverse-rotate opt, norms, opair contraction, assemble feats.
// ---------------------------------------------------------------------------
__global__ void finalize_kernel(const float* __restrict__ oacc,
                                const float* __restrict__ rotp,
                                const float* __restrict__ transp,
                                const float* __restrict__ z1df,
                                float* __restrict__ feats)
{
    const int i = blockIdx.x, tid = threadIdx.x;   // 256 threads
    float* fr = feats + (size_t)i * FEATW;

    for (int idx = tid; idx < HC_; idx += 256) {
        int h = idx >> 7, c = idx & 127;
        fr[idx] = oacc[((size_t)h * N_ + i) * VCW + c];
    }
    const float* Rm = rotp + (size_t)i * 9;
    const float tx = transp[i * 3], ty = transp[i * 3 + 1], tz = transp[i * 3 + 2];
    for (int idx = tid; idx < H_ * PV_; idx += 256) {
        int h = idx / PV_, p = idx % PV_;
        const float* ob = oacc + ((size_t)h * N_ + i) * VCW + 128 + p * 3;
        float dx = ob[0] - tx, dy = ob[1] - ty, dz = ob[2] - tz;
        float lx = Rm[0] * dx + Rm[3] * dy + Rm[6] * dz;   // rot^T * d
        float ly = Rm[1] * dx + Rm[4] * dy + Rm[7] * dz;
        float lz = Rm[2] * dx + Rm[5] * dy + Rm[8] * dz;
        fr[HC_ + idx]       = lx;
        fr[HC_ + 96 + idx]  = ly;
        fr[HC_ + 192 + idx] = lz;
        fr[HC_ + 288 + idx] = sqrtf(lx * lx + ly * ly + lz * lz + 1e-8f);
    }
    for (int idx = tid; idx < H_ * CZ4_; idx += 256) {
        int h = idx >> 5, c = idx & 31;
        const float* ob = oacc + ((size_t)h * N_ + i) * VCW + 176;
        fr[HC_ + 384 + idx] =
            z1df[((size_t)i * 2 + 0) * CZ4_ + c] * ob[c] +
            z1df[((size_t)i * 2 + 1) * CZ4_ + c] * ob[32 + c];
    }
}

// ---------------------------------------------------------------------------
extern "C" void kernel_launch(void* const* d_in, const int* in_sizes, int n_in,
                              void* d_out, int out_size, void* d_ws, size_t ws_size,
                              hipStream_t stream)
{
    const float* s     = (const float*)d_in[0];
    const float* z1    = (const float*)d_in[1];
    const float* z2    = (const float*)d_in[2];
    const float* rot   = (const float*)d_in[3];
    const float* trans = (const float*)d_in[4];
    const unsigned char* mask = (const unsigned char*)d_in[5];
    const float* w_q   = (const float*)d_in[6];
    const float* b_q   = (const float*)d_in[7];
    const float* w_kv  = (const float*)d_in[8];
    const float* b_kv  = (const float*)d_in[9];
    const float* w_qp  = (const float*)d_in[10];
    const float* b_qp  = (const float*)d_in[11];
    const float* w_kvp = (const float*)d_in[12];
    const float* b_kvp = (const float*)d_in[13];
    const float* w_b   = (const float*)d_in[14];
    const float* b_b   = (const float*)d_in[15];
    const float* w_dz  = (const float*)d_in[16];
    const float* b_dz  = (const float*)d_in[17];
    const float* hw    = (const float*)d_in[18];
    const float* w_out = (const float*)d_in[19];
    const float* b_out = (const float*)d_in[20];

    char* ws = (char*)d_ws;
    float*  qf   = (float*)(ws + OFF_QF);
    float*  kvf  = (float*)(ws + OFF_KVF);
    float*  qpf  = (float*)(ws + OFF_QPF);
    float*  kvpf = (float*)(ws + OFF_KVPF);
    float*  b1f  = (float*)(ws + OFF_B1F);
    float*  b2f  = (float*)(ws + OFF_B2F);
    float*  z1df = (float*)(ws + OFF_Z1DF);
    float*  z2df = (float*)(ws + OFF_Z2DF);
    __bf16* qbf  = (__bf16*)(ws + OFF_QBF);
    __bf16* kbf  = (__bf16*)(ws + OFF_KBF);
    __bf16* qabf = (__bf16*)(ws + OFF_QABF);
    __bf16* kabf = (__bf16*)(ws + OFF_KABF);
    __bf16* vcbf = (__bf16*)(ws + OFF_VCBF);
    float*  oacc = (float*)(ws + OFF_OACC);
    float*  feats= (float*)(ws + OFF_FEAT);

    const dim3 blk(128);
    auto gemm = [&](const float* X, const float* W, const float* bias, float* Y,
                    int nrows, int K, int M) {
        dim3 grid((M + 63) / 64, nrows / 64);
        hipLaunchKernelGGL(gemm_xwT_bf16, grid, blk, 0, stream, X, W, bias, Y,
                           nrows, K, M);
    };

    gemm(s,  w_q,   b_q,   qf,   N_,      CS_, HC_);
    gemm(s,  w_kv,  b_kv,  kvf,  N_,      CS_, 2 * HC_);
    gemm(s,  w_qp,  b_qp,  qpf,  N_,      CS_, H_ * PQ_ * 3);
    gemm(s,  w_kvp, b_kvp, kvpf, N_,      CS_, H_ * (PQ_ + PV_) * 3);
    gemm(z1, w_b,   b_b,   b1f,  N_ * R_, CZ_, H_);
    gemm(z2, w_b,   b_b,   b2f,  N_ * R_, CZ_, H_);
    gemm(z1, w_dz,  b_dz,  z1df, N_ * R_, CZ_, CZ4_);
    gemm(z2, w_dz,  b_dz,  z2df, N_ * R_, CZ_, CZ4_);

    hipLaunchKernelGGL(pack_kernel, dim3(N_), dim3(128), 0, stream,
                       qf, kvf, qpf, kvpf, b1f, b2f, z2df, rot, trans, mask, hw,
                       qbf, kbf, qabf, kabf, vcbf);

    hipLaunchKernelGGL(flash_ipa_kernel, dim3(N_ / 64, H_), dim3(128), 0, stream,
                       qbf, kbf, qabf, kabf, vcbf, oacc);

    hipLaunchKernelGGL(finalize_kernel, dim3(N_), dim3(256), 0, stream,
                       oacc, rot, trans, z1df, feats);
    gemm(feats, w_out, b_out, (float*)d_out, N_, FEATW, CS_);
}
